// SwinTransformerBlock_64922725646401
// MI455X (gfx1250) — matrix operand order repaired and checked by reading, not compiled, not run on hardware
//
#include <hip/hip_runtime.h>
#include <hip/hip_bf16.h>

typedef __attribute__((ext_vector_type(16))) _Float16 v16h;
typedef __attribute__((ext_vector_type(8)))  _Float16 v8h;
typedef __attribute__((ext_vector_type(8)))  float    v8f;

#define DEVINL __device__ __forceinline__

// ---------------- constants ----------------
constexpr int   Bsz   = 8;
constexpr int   Himg  = 224;
constexpr int   Cch   = 96;
constexpr int   WS_   = 7;
constexpr int   SS_   = 3;
constexpr int   NH_   = 3;
constexpr int   HD_   = 32;          // head dim
constexpr int   NTOK  = 49;          // tokens per window
constexpr int   NPAD  = 64;          // padded rows per window
constexpr int   NWX   = Himg / WS_;  // 32 windows per axis
constexpr int   NW1   = NWX * NWX;   // 1024 windows per image
constexpr int   NWIN  = Bsz * NW1;   // 8192 windows total
constexpr int   LTOK  = Himg * Himg; // 50176 tokens per image
constexpr int   MROW  = Bsz * LTOK;  // 401408 rows for MLP
constexpr float QSCALE = 0.17677669529663687f; // 1/sqrt(32)

// scratch layout (in halves)
constexpr size_t XW_E   = (size_t)NWIN * NPAD * Cch;        // 50,331,648
constexpr size_t LN2_E  = (size_t)MROW * Cch;               // 38,535,168
constexpr size_t WOFF   = 4 * XW_E;                         // f16 weights region
constexpr size_t WQ_E   = 288 * 96;
constexpr size_t WP_E   = 96 * 96;
constexpr size_t W1_E   = 384 * 96;
constexpr size_t W2_E   = 96 * 384;
// phase1: [xw | q | k | vT]   phase2 (same region): [ln2h | hidden]   then [weights]

// ---------------- WMMA helpers ----------------
DEVINL v8f wmma_f16(v16h a, v16h b, v8f c) {
  return __builtin_amdgcn_wmma_f32_16x16x32_f16(false, a, false, b, (short)0, c,
                                                false, false);
}

DEVINL v16h combine16(v8h lo, v8h hh) {
  v16h a;
#pragma unroll
  for (int t = 0; t < 8; ++t) { a[t] = lo[t]; a[t + 8] = hh[t]; }
  return a;
}

// A fragment, 16x32 f16 row-major. lane&15 = M row; per lane the 16 halves are
// contiguous: [k0+hi*16, +16)
DEVINL v16h frag_a(const _Float16* base, int ld, int row0, int k0, int lane) {
  int m = lane & 15, hi = lane >> 4;
  const _Float16* p = base + (size_t)(row0 + m) * ld + k0 + ((lane >> 4) << 4);
  (void)hi;
  v8h lo = *(const v8h*)p;
  v8h hh = *(const v8h*)(p + 8);
  return combine16(lo, hh);
}

// B fragment, 32x16 where B(k,n) = W[n][k], W f16 row-major [N][K].
// lane&15 = N col; lanes 0-15 hold K=k0..k0+15, lanes 16-31 hold K=k0+16..k0+31
// -> 32 contiguous bytes per lane.
DEVINL v16h frag_b16(const _Float16* W, int ldk, int n0, int k0, int lane) {
  int n = lane & 15, hi = lane >> 4;
  const _Float16* p = W + (size_t)(n0 + n) * ldk + k0 + (hi << 4);
  v8h lo = *(const v8h*)p;
  v8h hh = *(const v8h*)(p + 8);
  return combine16(lo, hh);
}

DEVINL int reg224(int c) { return c < (Himg - WS_) ? 0 : (c < (Himg - SS_) ? 1 : 2); }

// ---------------- K0: fp32 -> f16 weight conversion (runs once per call) --------
__global__ __launch_bounds__(256) void k_cvt(const float* __restrict__ a,
                                             _Float16* __restrict__ o, int n) {
  int i = blockIdx.x * 256 + threadIdx.x;
  if (i < n) o[i] = (_Float16)a[i];
}

// ---------------- K1: LN1 + roll(-3,-3) + window partition -> fp16 ----------------
__global__ __launch_bounds__(64) void k_ln1_part(const float* __restrict__ x,
                                                 const float* __restrict__ w,
                                                 const float* __restrict__ b,
                                                 _Float16* __restrict__ xw) {
  int win = blockIdx.x, t = threadIdx.x;
  _Float16* dst = xw + (size_t)win * NPAD * Cch + (size_t)t * Cch;
  if (t >= NTOK) {
#pragma unroll 4
    for (int c = 0; c < Cch; ++c) dst[c] = (_Float16)0.f;
    return;
  }
  int bi = win >> 10, wl = win & (NW1 - 1), wr = wl >> 5, wc = wl & 31;
  int i = t / WS_, j = t % WS_;
  int h0 = (wr * WS_ + i + SS_) % Himg;
  int w0 = (wc * WS_ + j + SS_) % Himg;
  const float* src = x + ((size_t)bi * LTOK + (size_t)h0 * Himg + w0) * Cch;
  float mu = 0.f, m2 = 0.f;
  for (int c = 0; c < Cch; ++c) { float v = src[c]; mu += v; m2 += v * v; }
  mu *= (1.f / Cch);
  float var = m2 * (1.f / Cch) - mu * mu;
  float inv = rsqrtf(var + 1e-5f);
  for (int c = 0; c < Cch; ++c)
    dst[c] = (_Float16)(((src[c] - mu) * inv) * w[c] + b[c]);
}

// ---------------- K2: QKV GEMM (per window 64x288x96), split + scale -----------
// v is stored TRANSPOSED: vT[win][head][dim][token] for contiguous B-fragments.
__global__ __launch_bounds__(256) void k_qkv(const _Float16* __restrict__ xw,
                                             const _Float16* __restrict__ Wq, // [288][96] f16
                                             const float* __restrict__ bias,
                                             _Float16* __restrict__ qb,
                                             _Float16* __restrict__ kb,
                                             _Float16* __restrict__ vtb) {
  int win = blockIdx.x;
  int wave = threadIdx.x >> 5, lane = threadIdx.x & 31;
  const _Float16* A = xw + (size_t)win * NPAD * Cch;
  for (int nt = wave; nt < 18; nt += 8) {
    v16h b0 = frag_b16(Wq, Cch, nt * 16, 0, lane);
    v16h b1 = frag_b16(Wq, Cch, nt * 16, 32, lane);
    v16h b2 = frag_b16(Wq, Cch, nt * 16, 64, lane);
#pragma unroll
    for (int mt = 0; mt < 4; ++mt) {
      v8f acc = {};
      acc = wmma_f16(frag_a(A, Cch, mt * 16, 0, lane), b0, acc);
      acc = wmma_f16(frag_a(A, Cch, mt * 16, 32, lane), b1, acc);
      acc = wmma_f16(frag_a(A, Cch, mt * 16, 64, lane), b2, acc);
      int n = lane & 15, hi = lane >> 4;
      int ncol = nt * 16 + n;
      float bv = bias[ncol];
      int sec = ncol / Cch;      // 0=q 1=k 2=v (tiles never straddle sections)
      int c = ncol - sec * Cch;
      int head = c >> 5, dim = c & 31;
      size_t hb = ((size_t)win * NH_ + head) * NPAD * HD_;
      if (sec == 2) {
        // transposed store: vT[dim][token], tokens contiguous
        _Float16* dst = vtb + hb + (size_t)dim * NPAD + mt * 16 + (hi << 3);
#pragma unroll
        for (int r = 0; r < 8; ++r) dst[r] = (_Float16)(acc[r] + bv);
      } else {
        _Float16* dst = ((sec == 0) ? qb : kb) + hb + dim;
        float sc = (sec == 0) ? QSCALE : 1.0f;
#pragma unroll
        for (int r = 0; r < 8; ++r) {
          int row = mt * 16 + r + (hi << 3);
          dst[(size_t)row * HD_] = (_Float16)((acc[r] + bv) * sc);
        }
      }
    }
  }
}

// ---------------- K3: attention, one wave per (window, head) ----------------
__global__ __launch_bounds__(64) void k_attn(const _Float16* __restrict__ qb,
                                             const _Float16* __restrict__ kb,
                                             const _Float16* __restrict__ vtb,
                                             const float* __restrict__ rpb, // [169][3]
                                             _Float16* __restrict__ outb) {
  __shared__ float    S[2][NPAD * NPAD];
  __shared__ _Float16 P[2][NPAD * NPAD];
  int wv = threadIdx.x >> 5, lane = threadIdx.x & 31;
  int task = blockIdx.x * 2 + wv;        // < 24576
  int win = task / NH_, head = task - win * NH_;
  size_t qoff = (size_t)task * NPAD * HD_;  // task == win*NH_+head, matches layout
  const _Float16* q  = qb  + qoff;
  const _Float16* k  = kb  + qoff;
  const _Float16* vt = vtb + qoff;
  int wl = win & (NW1 - 1), wr = wl >> 5, wc = wl & 31;

  // S = q * k^T  (K = head dim = 32 -> single MMA per 16x16 tile)
#pragma unroll
  for (int mt = 0; mt < 4; ++mt) {
#pragma unroll
    for (int nt = 0; nt < 4; ++nt) {
      v8f acc = {};
      acc = wmma_f16(frag_a(q, HD_, mt * 16, 0, lane),
                     frag_b16(k, HD_, nt * 16, 0, lane), acc);
      int n = lane & 15, hi = lane >> 4;
#pragma unroll
      for (int r = 0; r < 8; ++r)
        S[wv][(size_t)(mt * 16 + r + (hi << 3)) * NPAD + nt * 16 + n] = acc[r];
    }
  }
  __syncthreads();

  // softmax + rpb + shift-window mask; lane handles rows lane and lane+32
#pragma unroll
  for (int mm = 0; mm < 2; ++mm) {
    int m = lane + 32 * mm;
    float* srow = &S[wv][(size_t)m * NPAD];
    int mc = (m < NTOK) ? m : NTOK - 1;
    int mi = mc / WS_, mj = mc % WS_;
    int rid_m = reg224(wr * WS_ + mi) * 3 + reg224(wc * WS_ + mj);
    float mx = -1e30f;
    for (int n = 0; n < NPAD; ++n) {
      float s;
      if (n < NTOK) {
        int ni = n / WS_, nj = n % WS_;
        int rid_n = reg224(wr * WS_ + ni) * 3 + reg224(wc * WS_ + nj);
        float rb = rpb[((mi - ni + WS_ - 1) * (2 * WS_ - 1) + (mj - nj + WS_ - 1)) * NH_ + head];
        s = srow[n] + rb + ((rid_n != rid_m) ? -100.0f : 0.0f);
      } else {
        s = -1e30f;   // padded keys excluded
      }
      srow[n] = s;
      mx = fmaxf(mx, s);
    }
    float sum = 0.f;
    for (int n = 0; n < NPAD; ++n) {
      float e = __expf(srow[n] - mx);
      sum += e;
      srow[n] = e;
    }
    float inv = 1.0f / sum;
    _Float16* prow = &P[wv][(size_t)m * NPAD];
    for (int n = 0; n < NPAD; ++n) prow[n] = (_Float16)(srow[n] * inv);
  }
  __syncthreads();

  // O = P * v ; B(k,n) = V[token k][dim n] = vT[n][k] -> contiguous fragments
  const _Float16* Pw = &P[wv][0];
#pragma unroll
  for (int mt = 0; mt < 4; ++mt) {
#pragma unroll
    for (int nt = 0; nt < 2; ++nt) {
      v8f acc = {};
      acc = wmma_f16(frag_a(Pw, NPAD, mt * 16, 0, lane),
                     frag_b16(vt, NPAD, nt * 16, 0, lane), acc);
      acc = wmma_f16(frag_a(Pw, NPAD, mt * 16, 32, lane),
                     frag_b16(vt, NPAD, nt * 16, 32, lane), acc);
      int n = lane & 15, hi = lane >> 4;
#pragma unroll
      for (int r = 0; r < 8; ++r) {
        int row = mt * 16 + r + (hi << 3);
        if (row < NTOK)
          outb[((size_t)win * NPAD + row) * Cch + head * HD_ + nt * 16 + n] =
              (_Float16)acc[r];
      }
    }
  }
}

// ---------------- K4: proj GEMM + residual + window-reverse + unshift ----------------
__global__ __launch_bounds__(256) void k_proj(const _Float16* __restrict__ ao,
                                              const _Float16* __restrict__ Wp, // [96][96] f16
                                              const float* __restrict__ bias,
                                              const float* __restrict__ x,
                                              float* __restrict__ out) {
  int win = blockIdx.x;
  int wave = threadIdx.x >> 5, lane = threadIdx.x & 31;
  const _Float16* A = ao + (size_t)win * NPAD * Cch;
  int bi = win >> 10, wl = win & (NW1 - 1), wr = wl >> 5, wc = wl & 31;
  for (int t = wave; t < 24; t += 8) {
    int mt = t / 6, nt = t - (t / 6) * 6;
    v8f acc = {};
#pragma unroll
    for (int ks = 0; ks < 3; ++ks)
      acc = wmma_f16(frag_a(A, Cch, mt * 16, ks * 32, lane),
                     frag_b16(Wp, Cch, nt * 16, ks * 32, lane), acc);
    int n = lane & 15, hi = lane >> 4;
    int ncol = nt * 16 + n;
    float bv = bias[ncol];
#pragma unroll
    for (int r = 0; r < 8; ++r) {
      int row = mt * 16 + r + (hi << 3);
      if (row < NTOK) {
        int i = row / WS_, j = row % WS_;
        int h0 = (wr * WS_ + i + SS_) % Himg;
        int w0 = (wc * WS_ + j + SS_) % Himg;
        size_t gi = ((size_t)bi * LTOK + (size_t)h0 * Himg + w0) * Cch + ncol;
        out[gi] = x[gi] + acc[r] + bv;   // x2 = shortcut + attn_proj
      }
    }
  }
}

// ---------------- K5: LN2 -> fp16 ----------------
__global__ __launch_bounds__(64) void k_ln2(const float* __restrict__ x2,
                                            const float* __restrict__ w,
                                            const float* __restrict__ b,
                                            _Float16* __restrict__ y) {
  size_t row = (size_t)blockIdx.x * 64 + threadIdx.x;
  const float* src = x2 + row * Cch;
  _Float16* dst = y + row * Cch;
  float mu = 0.f, m2 = 0.f;
  for (int c = 0; c < Cch; ++c) { float v = src[c]; mu += v; m2 += v * v; }
  mu *= (1.f / Cch);
  float var = m2 * (1.f / Cch) - mu * mu;
  float inv = rsqrtf(var + 1e-5f);
  for (int c = 0; c < Cch; ++c)
    dst[c] = (_Float16)(((src[c] - mu) * inv) * w[c] + b[c]);
}

// ---------------- K6: FC1 GEMM (64 x 384 x 96 per WG) + exact GELU ----------------
__global__ __launch_bounds__(256) void k_fc1(const _Float16* __restrict__ a,
                                             const _Float16* __restrict__ W, // [384][96] f16
                                             const float* __restrict__ bias,
                                             _Float16* __restrict__ h) {
  size_t rowbase = (size_t)blockIdx.x * 64;
  int wave = threadIdx.x >> 5, lane = threadIdx.x & 31;
  const _Float16* A = a + rowbase * Cch;
  for (int nt = wave; nt < 24; nt += 8) {
    v16h b0 = frag_b16(W, Cch, nt * 16, 0, lane);
    v16h b1 = frag_b16(W, Cch, nt * 16, 32, lane);
    v16h b2 = frag_b16(W, Cch, nt * 16, 64, lane);
#pragma unroll
    for (int mt = 0; mt < 4; ++mt) {
      v8f acc = {};
      acc = wmma_f16(frag_a(A, Cch, mt * 16, 0, lane), b0, acc);
      acc = wmma_f16(frag_a(A, Cch, mt * 16, 32, lane), b1, acc);
      acc = wmma_f16(frag_a(A, Cch, mt * 16, 64, lane), b2, acc);
      int n = lane & 15, hi = lane >> 4;
      int ncol = nt * 16 + n;
      float bv = bias[ncol];
#pragma unroll
      for (int r = 0; r < 8; ++r) {
        int row = mt * 16 + r + (hi << 3);
        float v = acc[r] + bv;
        float g = 0.5f * v * (1.0f + erff(v * 0.70710678118654752f));
        h[(rowbase + row) * 384 + ncol] = (_Float16)g;
      }
    }
  }
}

// ---------------- K7: FC2 GEMM (64 x 96 x 384 per WG), out += m ----------------
__global__ __launch_bounds__(256) void k_fc2(const _Float16* __restrict__ hbuf,
                                             const _Float16* __restrict__ W, // [96][384] f16
                                             const float* __restrict__ bias,
                                             float* __restrict__ out) {
  size_t rowbase = (size_t)blockIdx.x * 64;
  int wave = threadIdx.x >> 5, lane = threadIdx.x & 31;
  const _Float16* A = hbuf + rowbase * 384;
  for (int t = wave; t < 24; t += 8) {
    int mt = t / 6, nt = t - (t / 6) * 6;
    v8f acc = {};
    for (int k0 = 0; k0 < 384; k0 += 32)
      acc = wmma_f16(frag_a(A, 384, mt * 16, k0, lane),
                     frag_b16(W, 384, nt * 16, k0, lane), acc);
    int n = lane & 15, hi = lane >> 4;
    int ncol = nt * 16 + n;
    float bv = bias[ncol];
#pragma unroll
    for (int r = 0; r < 8; ++r) {
      int row = mt * 16 + r + (hi << 3);
      float* o = out + (rowbase + row) * Cch + ncol;
      *o = *o + acc[r] + bv;   // final = x2 + mlp
    }
  }
}

// ---------------- launch ----------------
extern "C" void kernel_launch(void* const* d_in, const int* in_sizes, int n_in,
                              void* d_out, int out_size, void* d_ws, size_t ws_size,
                              hipStream_t stream) {
  (void)in_sizes; (void)n_in; (void)out_size; (void)ws_size;
  const float* x      = (const float*)d_in[0];
  const float* ln1_w  = (const float*)d_in[1];
  const float* ln1_b  = (const float*)d_in[2];
  const float* qkv_w  = (const float*)d_in[3];
  const float* qkv_b  = (const float*)d_in[4];
  const float* proj_w = (const float*)d_in[5];
  const float* proj_b = (const float*)d_in[6];
  const float* rpb    = (const float*)d_in[7];
  const float* ln2_w  = (const float*)d_in[8];
  const float* ln2_b  = (const float*)d_in[9];
  const float* fc1_w  = (const float*)d_in[10];
  const float* fc1_b  = (const float*)d_in[11];
  const float* fc2_w  = (const float*)d_in[12];
  const float* fc2_b  = (const float*)d_in[13];

  _Float16* hws = (_Float16*)d_ws;
  _Float16* xw  = hws;                 // phase1: LN'd windows; later: attn output
  _Float16* qb  = hws + XW_E;
  _Float16* kb  = hws + 2 * XW_E;
  _Float16* vtb = hws + 3 * XW_E;      // V transposed
  _Float16* ln2h = hws;                // phase2 reuse
  _Float16* hid  = hws + LN2_E;
  _Float16* wq16 = hws + WOFF;
  _Float16* wp16 = wq16 + WQ_E;
  _Float16* w116 = wp16 + WP_E;
  _Float16* w216 = w116 + W1_E;
  float* out = (float*)d_out;

  k_cvt<<<(int)((WQ_E + 255) / 256), 256, 0, stream>>>(qkv_w,  wq16, (int)WQ_E);
  k_cvt<<<(int)((WP_E + 255) / 256), 256, 0, stream>>>(proj_w, wp16, (int)WP_E);
  k_cvt<<<(int)((W1_E + 255) / 256), 256, 0, stream>>>(fc1_w,  w116, (int)W1_E);
  k_cvt<<<(int)((W2_E + 255) / 256), 256, 0, stream>>>(fc2_w,  w216, (int)W2_E);

  k_ln1_part<<<NWIN, 64, 0, stream>>>(x, ln1_w, ln1_b, xw);
  k_qkv<<<NWIN, 256, 0, stream>>>(xw, wq16, qkv_b, qb, kb, vtb);
  k_attn<<<(NWIN * NH_) / 2, 64, 0, stream>>>(qb, kb, vtb, rpb, xw);
  k_proj<<<NWIN, 256, 0, stream>>>(xw, wp16, proj_b, x, out);
  k_ln2<<<MROW / 64, 64, 0, stream>>>(out, ln2_w, ln2_b, ln2h);
  k_fc1<<<MROW / 64, 256, 0, stream>>>(ln2h, w116, fc1_b, hid);
  k_fc2<<<MROW / 64, 256, 0, stream>>>(hid, w216, fc2_b, out);
}
